// KVMN_42099269435998
// MI455X (gfx1250) — matrix operand order repaired and checked by reading, not compiled, not run on hardware
//
#include <hip/hip_runtime.h>
#include <hip/hip_bf16.h>

#define BB   256
#define TT   50
#define VV   100000
#define EMB  64
#define KBD  64
#define HH   128
#define NF   10
#define OUTD 64

typedef __attribute__((ext_vector_type(16))) _Float16 v16h;
typedef __attribute__((ext_vector_type(8)))  float    v8f;
typedef __attribute__((ext_vector_type(4)))  float    v4f;

__device__ __forceinline__ float sig_(float x) { return 1.0f / (1.0f + __expf(-x)); }

// ---- WMMA f32 = f16 x f16 + f32, 16x16x32 -------------------------------
__device__ __forceinline__ v8f wmma32(v16h a, v16h b, v8f c) {
    return __builtin_amdgcn_wmma_f32_16x16x32_f16(
        /*neg_a=*/false, a, /*neg_b=*/false, b,
        /*c_mod=*/(short)0, c, /*reuse_a=*/false, /*reuse_b=*/false);
}

// A fragment (16 x 32) from f16 LDS, ISA 16-bit A K-interleave:
//   halves 0..7  -> K = kbase + (lane>>4)*8 + j        (16B contiguous)
//   halves 8..15 -> K = kbase + (lane>>4)*8 + 16 + j-8 (16B contiguous)
__device__ __forceinline__ v16h ldsAh(const _Float16* X, int ld, int kbase) {
    const int lane = threadIdx.x & 31;
    const _Float16* r = X + (lane & 15) * ld + kbase + ((lane >> 4) << 3);
    v16h a;
#pragma unroll
    for (int j = 0; j < 8; ++j) { a[j] = r[j]; a[j + 8] = r[j + 16]; }
    return a;
}
__device__ __forceinline__ v16h gAh(const _Float16* X, int ld, int m0, int kbase) {
    const int lane = threadIdx.x & 31;
    const _Float16* r = X + (size_t)(m0 + (lane & 15)) * ld + kbase + ((lane >> 4) << 3);
    v16h a;
#pragma unroll
    for (int j = 0; j < 8; ++j) { a[j] = r[j]; a[j + 8] = r[j + 16]; }
    return a;
}
// B fragment (32 x 16): col n = n0 + (lane&15); K interleaved like A (8h+j, 16+8h+j-8).
__device__ __forceinline__ v16h gB(const _Float16* W, int ld, int n0, int kbase) {
    const int lane = threadIdx.x & 31;
    const _Float16* r = W + (size_t)(n0 + (lane & 15)) * ld + kbase + ((lane >> 4) << 3);
    v16h b;
#pragma unroll
    for (int j = 0; j < 8; ++j) { b[j] = r[j]; b[j + 8] = r[j + 16]; }
    return b;
}
__device__ __forceinline__ v16h ldsBh(const _Float16* S, int ld, int n0, int kbase) {
    const int lane = threadIdx.x & 31;
    const _Float16* r = S + (n0 + (lane & 15)) * ld + kbase + ((lane >> 4) << 3);
    v16h b;
#pragma unroll
    for (int j = 0; j < 8; ++j) { b[j] = r[j]; b[j + 8] = r[j + 16]; }
    return b;
}
// C/D tile stores: VGPR r, lane L -> row (L>>4)*8 + r, col n0 + (L&15)
__device__ __forceinline__ void stTile(float* Y, int ld, int n0, v8f acc) {
    const int lane = threadIdx.x & 31;
    const int n = n0 + (lane & 15), h = lane >> 4;
#pragma unroll
    for (int r = 0; r < 8; ++r) Y[(h * 8 + r) * ld + n] = acc[r];
}
__device__ __forceinline__ void stTileH(_Float16* Y, int ld, int n0, v8f acc) {
    const int lane = threadIdx.x & 31;
    const int n = n0 + (lane & 15), h = lane >> 4;
#pragma unroll
    for (int r = 0; r < 8; ++r) Y[(h * 8 + r) * ld + n] = (_Float16)acc[r];
}

// ---- weight f32 -> f16 ---------------------------------------------------
__global__ void f32_to_f16(const float* __restrict__ src, _Float16* __restrict__ dst, int n) {
    int i = blockIdx.x * 256 + threadIdx.x;
    if (i < n) dst[i] = (_Float16)src[i];
}

// ---- recurrent phase: 16 blocks x 16 batch rows, all state in LDS --------
__global__ __launch_bounds__(128, 1)
void kvmn_recurrent(const int* __restrict__ X,
                    const float* __restrict__ E, const float* __restrict__ KBE,
                    const float* __restrict__ rmat,
                    const float* __restrict__ bih, const float* __restrict__ bhh,
                    const float* __restrict__ u_b, const float* __restrict__ ea_b,
                    const float* __restrict__ u2_b, const float* __restrict__ m1_b,
                    const _Float16* __restrict__ Wih, const _Float16* __restrict__ Whh,
                    const _Float16* __restrict__ uW,  const _Float16* __restrict__ eaW,
                    const _Float16* __restrict__ u2W, const _Float16* __restrict__ m1W,
                    _Float16* __restrict__ yact) {
    extern __shared__ char smemraw[];
    // f32 regions (elementwise consumers)
    float* sh_gi = (float*)(smemraw);                //  16*384          @      0
    float* sh_gh = (float*)(smemraw + 24576);        //  16*384
    float* sh_MN = (float*)(smemraw + 49152);        //  16*10*64
    float* sh_hf = (float*)(smemraw + 90112);        //  16*128  h (f32)
    float* sh_EA = (float*)(smemraw + 98304);        //  16*64
    float* sh_U  = (float*)(smemraw + 102400);       //  16*64
    float* sh_ys = (float*)(smemraw + 106496);       //  16*64
    float* sh_AW = (float*)(smemraw + 110592);       //  16*16
    float* sh_rm = (float*)(smemraw + 111616);       //  10*64 r_matrix
    // f16 regions (WMMA A-operand staging)
    _Float16* sh_xh   = (_Float16*)(smemraw + 114176); // 16*64
    _Float16* sh_kbh  = (_Float16*)(smemraw + 116224); // 16*64
    _Float16* sh_hACh = (_Float16*)(smemraw + 118272); // 16*192 (h | AC)
    _Float16* sh_Uth  = (_Float16*)(smemraw + 124416); // 16*128
    int* sh_idx = (int*)(smemraw + 128512);            // 16

    const int tid  = threadIdx.x;
    const int wave = tid >> 5;
    const int lane = tid & 31;
    const int b0   = blockIdx.x * 16;

    for (int i = tid; i < 2048;  i += 128) sh_hf[i]   = 0.0f;
    for (int i = tid; i < 3072;  i += 128) sh_hACh[i] = (_Float16)0.0f;
    for (int i = tid; i < 10240; i += 128) sh_MN[i]   = 0.0f;
    for (int i = tid; i < 1024;  i += 128) sh_ys[i]   = 0.0f;
    for (int i = tid; i < NF * 64; i += 128) sh_rm[i] = rmat[i];
    __syncthreads();

    for (int t = 0; t < TT; ++t) {
        if (tid < 16) sh_idx[tid] = X[(b0 + tid) * TT + t];
        __syncthreads();
        for (int i = tid; i < 1024; i += 128) {
            int m = i >> 6, d = i & 63;
            int idx = sh_idx[m];
            sh_xh[i]  = (idx == 0) ? (_Float16)0.0f : (_Float16)E  [(size_t)idx * EMB + d];
            sh_kbh[i] = (idx == 0) ? (_Float16)0.0f : (_Float16)KBE[(size_t)idx * KBD + d];
        }
        __syncthreads();

        // --- hoisted A fragments (pure b128 LDS loads, no cvt) ---
        v16h ax0 = ldsAh(sh_xh, 64, 0), ax1 = ldsAh(sh_xh, 64, 32);
        v16h akb0 = ldsAh(sh_kbh, 64, 0), akb1 = ldsAh(sh_kbh, 64, 32);
        v16h ah[4];
#pragma unroll
        for (int k = 0; k < 4; ++k) ah[k] = ldsAh(sh_hACh, 192, k * 32);

        // GRU pre-activations: gi = x@Wih.T + bih ; gh = h@Whh.T + bhh  (24 N-tiles)
#pragma unroll
        for (int i = 0; i < 6; ++i) {
            int n0 = (wave + 4 * i) * 16;
            v8f gi = {};
            gi = wmma32(ax0, gB(Wih, 64, n0, 0),  gi);
            gi = wmma32(ax1, gB(Wih, 64, n0, 32), gi);
            v8f gh = {};
#pragma unroll
            for (int k = 0; k < 4; ++k)
                gh = wmma32(ah[k], gB(Whh, HH, n0, k * 32), gh);
            float bi = bih[n0 + (lane & 15)], bh = bhh[n0 + (lane & 15)];
#pragma unroll
            for (int r = 0; r < 8; ++r) { gi[r] += bi; gh[r] += bh; }
            stTile(sh_gi, 384, n0, gi);
            stTile(sh_gh, 384, n0, gh);
        }
        // EA = tanh(kb @ eaW.T + ea_b)    (1 tile per wave)
        {
            int n0 = wave * 16;
            v8f a = {};
            a = wmma32(akb0, gB(eaW, 64, n0, 0),  a);
            a = wmma32(akb1, gB(eaW, 64, n0, 32), a);
            float bb = ea_b[n0 + (lane & 15)];
#pragma unroll
            for (int r = 0; r < 8; ++r) a[r] = tanhf(a[r] + bb);
            stTile(sh_EA, 64, n0, a);
        }
        __syncthreads();

        // GRU cell elementwise: f32 update + f16 shadow for WMMA
        for (int i = tid; i < 16 * HH; i += 128) {
            int m = i >> 7, j = i & 127;
            const float* gi = sh_gi + m * 384;
            const float* gh = sh_gh + m * 384;
            float r = sig_(gi[j] + gh[j]);
            float z = sig_(gi[128 + j] + gh[128 + j]);
            float n = tanhf(gi[256 + j] + r * gh[256 + j]);
            float hn = (1.0f - z) * n + z * sh_hf[m * 128 + j];
            sh_hf[m * 128 + j] = hn;
            sh_hACh[m * 192 + j] = (_Float16)hn;
        }
        // memory write: gate = sig(<MN, EA+r>) ; MN = lerp
        for (int p = tid; p < 16 * NF; p += 128) {
            int m = p / NF, f = p % NF;
            float* MNp = sh_MN + (m * NF + f) * 64;
            const float* EAm = sh_EA + m * 64;
            const float* rf  = sh_rm + f * 64;
            float dot = 0.0f;
            for (int d = 0; d < 64; ++d) dot += MNp[d] * (EAm[d] + rf[d]);
            float g = sig_(dot);
            for (int d = 0; d < 64; ++d) {
                float ea = EAm[d] + rf[d];
                MNp[d] = MNp[d] * (1.0f - g) + ea * g;
            }
        }
        __syncthreads();

        // reload h fragments ONCE after the GRU update; reused by U and Ut GEMMs
        v16h ahn[4];
#pragma unroll
        for (int k = 0; k < 4; ++k) ahn[k] = ldsAh(sh_hACh, 192, k * 32);

        // U = tanh(h @ uW.T + u_b)
        {
            int n0 = wave * 16;
            v8f a = {};
#pragma unroll
            for (int k = 0; k < 4; ++k)
                a = wmma32(ahn[k], gB(uW, HH, n0, k * 32), a);
            float bb = u_b[n0 + (lane & 15)];
#pragma unroll
            for (int r = 0; r < 8; ++r) a[r] = tanhf(a[r] + bb);
            stTile(sh_U, 64, n0, a);
        }
        __syncthreads();
        // attention scores + softmax over NF
        for (int p = tid; p < 16 * NF; p += 128) {
            int m = p / NF, f = p % NF;
            const float* Um = sh_U + m * 64;
            const float* rf = sh_rm + f * 64;
            float dot = 0.0f;
            for (int d = 0; d < 64; ++d) dot += Um[d] * rf[d];
            sh_AW[m * 16 + f] = dot;
        }
        __syncthreads();
        if (tid < 16) {
            float mx = -1e30f;
            for (int f = 0; f < NF; ++f) mx = fmaxf(mx, sh_AW[tid * 16 + f]);
            float s = 0.0f;
            for (int f = 0; f < NF; ++f) { float e = __expf(sh_AW[tid * 16 + f] - mx); sh_AW[tid * 16 + f] = e; s += e; }
            float inv = 1.0f / s;
            for (int f = 0; f < NF; ++f) sh_AW[tid * 16 + f] *= inv;
        }
        __syncthreads();
        // AC = sum_f MN * AW  (WMMA-only consumer -> f16)
        for (int i = tid; i < 1024; i += 128) {
            int m = i >> 6, d = i & 63;
            float acc = 0.0f;
            for (int f = 0; f < NF; ++f) acc += sh_MN[(m * NF + f) * 64 + d] * sh_AW[m * 16 + f];
            sh_hACh[m * 192 + 128 + d] = (_Float16)acc;
        }
        __syncthreads();
        // Ut = tanh([h|AC] @ u2W.T + u2_b)  (8 tiles) — h fragments reused
        v16h aac0 = ldsAh(sh_hACh, 192, 128), aac1 = ldsAh(sh_hACh, 192, 160);
#pragma unroll
        for (int i = 0; i < 2; ++i) {
            int n0 = (wave + 4 * i) * 16;
            v8f a = {};
#pragma unroll
            for (int k = 0; k < 4; ++k)
                a = wmma32(ahn[k], gB(u2W, 192, n0, k * 32), a);
            a = wmma32(aac0, gB(u2W, 192, n0, 128), a);
            a = wmma32(aac1, gB(u2W, 192, n0, 160), a);
            float bb = u2_b[n0 + (lane & 15)];
#pragma unroll
            for (int r = 0; r < 8; ++r) a[r] = tanhf(a[r] + bb);
            stTileH(sh_Uth, HH, n0, a);
        }
        __syncthreads();
        // ysum += tanh(Ut @ m1W.T + m1_b)
        {
            v16h aut[4];
#pragma unroll
            for (int k = 0; k < 4; ++k) aut[k] = ldsAh(sh_Uth, HH, k * 32);
            int n0 = wave * 16;
            v8f a = {};
#pragma unroll
            for (int k = 0; k < 4; ++k)
                a = wmma32(aut[k], gB(m1W, HH, n0, k * 32), a);
            float bb = m1_b[n0 + (lane & 15)];
            int n = n0 + (lane & 15), h2 = lane >> 4;
#pragma unroll
            for (int r = 0; r < 8; ++r) sh_ys[(h2 * 8 + r) * 64 + n] += tanhf(a[r] + bb);
        }
        __syncthreads();
    }
    // write averaged y activations as f16 (consumed directly as WMMA A operand)
    for (int i = tid; i < 1024; i += 128)
        yact[b0 * OUTD + i] = (_Float16)(sh_ys[i] * (1.0f / (float)TT));
}

// ---- vocab phase: fused Sy + logits, NT stores ---------------------------
__global__ __launch_bounds__(128)
void kvmn_vocab(const float* __restrict__ E, const float* __restrict__ KBE,
                const _Float16* __restrict__ yact,
                const _Float16* __restrict__ m2W, const float* __restrict__ m2_b,
                float* __restrict__ out) {
    __shared__ _Float16 sh_Sy[4][16 * 64 + 32];
    const int tid = threadIdx.x, wave = tid >> 5, lane = tid & 31;
    const int v0 = (blockIdx.x * 4 + wave) * 16;
    if (v0 >= VV) return;

    // prefetch next tile's embedding rows into L2 (global_prefetch_b8)
    {
        int vp = v0 + 64 + (lane & 15);
        if (vp < VV) {
            __builtin_prefetch(E   + (size_t)vp * EMB, 0, 1);
            __builtin_prefetch(KBE + (size_t)vp * KBD, 0, 1);
        }
    }

    // Sy tile: [16 vocab x 64 out] = tanh(MergeE @ m2W.T + m2_b)
    v8f sy[4] = {{}, {}, {}, {}};
    const int m = lane & 15, h = lane >> 4;
    const int v = v0 + m;
    const bool valid = (v < VV) && (v != 0);
#pragma unroll
    for (int kc = 0; kc < 4; ++kc) {
        const int kbase = kc * 32;
        const float* tab = (kbase < 64) ? E : KBE;
        const float* r = tab + (size_t)v * 64 + (kbase & 63) + (h << 3);
        v16h a;
#pragma unroll
        for (int j = 0; j < 8; ++j) {
            a[j]     = valid ? (_Float16)r[j]      : (_Float16)0.0f;
            a[j + 8] = valid ? (_Float16)r[j + 16] : (_Float16)0.0f;
        }
#pragma unroll
        for (int nt = 0; nt < 4; ++nt)
            sy[nt] = wmma32(a, gB(m2W, 128, nt * 16, kbase), sy[nt]);
    }
    _Float16* Sy = sh_Sy[wave];
#pragma unroll
    for (int nt = 0; nt < 4; ++nt) {
        float bb = m2_b[nt * 16 + (lane & 15)];
        int n = nt * 16 + (lane & 15);
#pragma unroll
        for (int r = 0; r < 8; ++r) Sy[(h * 8 + r) * 64 + n] = (_Float16)tanhf(sy[nt][r] + bb);
    }
    // wave-private LDS; same-wave DS ops are in-order, compiler inserts waits

    // logits: out[b, v0..v0+15] = y @ Sy.T   (B operand = Sy rows, K = out dim)
    v16h bf0 = ldsBh(Sy, 64, 0, 0);
    v16h bf1 = ldsBh(Sy, 64, 0, 32);
    const int vcol = v0 + (lane & 15);
#pragma unroll
    for (int mt = 0; mt < BB / 16; ++mt) {
        v8f acc = {};
        acc = wmma32(gAh(yact, OUTD, mt * 16, 0),  bf0, acc);
        acc = wmma32(gAh(yact, OUTD, mt * 16, 32), bf1, acc);
        if (vcol < VV) {
#pragma unroll
            for (int r = 0; r < 8; ++r)
                __builtin_nontemporal_store(
                    acc[r], &out[(size_t)(mt * 16 + h * 8 + r) * VV + vcol]);
        }
    }
}

// ---- log-softmax over V (NT streaming; keep L2 for embeddings) -----------
__global__ __launch_bounds__(256)
void kvmn_rowstat(const float* __restrict__ out, float* __restrict__ corr) {
    __shared__ float red[256];
    const int b = blockIdx.x;
    const float* row = out + (size_t)b * VV;
    float mx = -3.4e38f;
    for (int v = threadIdx.x; v < VV; v += 256)
        mx = fmaxf(mx, __builtin_nontemporal_load(&row[v]));
    red[threadIdx.x] = mx; __syncthreads();
    for (int s = 128; s > 0; s >>= 1) {
        if (threadIdx.x < s) red[threadIdx.x] = fmaxf(red[threadIdx.x], red[threadIdx.x + s]);
        __syncthreads();
    }
    const float m = red[0]; __syncthreads();
    float se = 0.0f;
    for (int v = threadIdx.x; v < VV; v += 256)
        se += __expf(__builtin_nontemporal_load(&row[v]) - m);
    red[threadIdx.x] = se; __syncthreads();
    for (int s = 128; s > 0; s >>= 1) {
        if (threadIdx.x < s) red[threadIdx.x] += red[threadIdx.x + s];
        __syncthreads();
    }
    if (threadIdx.x == 0) corr[b] = m + __logf(red[0]);
}

__global__ __launch_bounds__(256)
void kvmn_logsoftmax(float* __restrict__ out, const float* __restrict__ corr) {
    const int b = blockIdx.y;
    const float c = corr[b];
    const size_t base = (size_t)b * VV;
    const int v = (blockIdx.x * 256 + threadIdx.x) * 4;
    if (v + 3 < VV) {
        v4f* p = (v4f*)(out + base + v);
        v4f x = __builtin_nontemporal_load(p);
        x[0] -= c; x[1] -= c; x[2] -= c; x[3] -= c;
        __builtin_nontemporal_store(x, p);
    } else {
        for (int i = v; i < VV && i < v + 4; ++i) out[base + i] -= c;
    }
}

extern "C" void kernel_launch(void* const* d_in, const int* in_sizes, int n_in,
                              void* d_out, int out_size, void* d_ws, size_t ws_size,
                              hipStream_t stream) {
    (void)in_sizes; (void)n_in; (void)out_size; (void)ws_size;
    const int*   X    = (const int*)  d_in[0];
    const float* E    = (const float*)d_in[1];
    const float* KBE  = (const float*)d_in[2];
    const float* rmat = (const float*)d_in[3];
    const float* Wih  = (const float*)d_in[4];
    const float* Whh  = (const float*)d_in[5];
    const float* bih  = (const float*)d_in[6];
    const float* bhh  = (const float*)d_in[7];
    const float* uW   = (const float*)d_in[8];
    const float* u_b  = (const float*)d_in[9];
    const float* eaW  = (const float*)d_in[10];
    const float* ea_b = (const float*)d_in[11];
    const float* u2W  = (const float*)d_in[12];
    const float* u2_b = (const float*)d_in[13];
    const float* m1W  = (const float*)d_in[14];
    const float* m1_b = (const float*)d_in[15];
    const float* m2W  = (const float*)d_in[16];
    const float* m2_b = (const float*)d_in[17];
    float* out = (float*)d_out;

    char* ws = (char*)d_ws;
    _Float16* Wih_h = (_Float16*)ws;
    _Float16* Whh_h = Wih_h + 384 * 64;
    _Float16* uW_h  = Whh_h + 384 * 128;
    _Float16* eaW_h = uW_h  + 64 * 128;
    _Float16* u2W_h = eaW_h + 64 * 64;
    _Float16* m1W_h = u2W_h + 128 * 192;
    _Float16* m2W_h = m1W_h + 64 * 128;
    _Float16* yact  = (_Float16*)(ws + 262144);     // 256*64 f16
    float*    corr  = (float*)(ws + 262144 + BB * OUTD * sizeof(_Float16));

    auto cv = [&](const float* s, _Float16* d, int n) {
        f32_to_f16<<<(n + 255) / 256, 256, 0, stream>>>(s, d, n);
    };
    cv(Wih, Wih_h, 384 * 64);
    cv(Whh, Whh_h, 384 * 128);
    cv(uW,  uW_h,  64 * 128);
    cv(eaW, eaW_h, 64 * 64);
    cv(u2W, u2W_h, 128 * 192);
    cv(m1W, m1W_h, 64 * 128);
    cv(m2W, m2W_h, 64 * 128);

    const size_t smemBytes = 128576; // ~126 KB of 320 KB LDS
    kvmn_recurrent<<<BB / 16, 128, smemBytes, stream>>>(
        X, E, KBE, rmat, bih, bhh, u_b, ea_b, u2_b, m1_b,
        Wih_h, Whh_h, uW_h, eaW_h, u2W_h, m1W_h, yact);

    kvmn_vocab<<<(VV + 63) / 64, 128, 0, stream>>>(E, KBE, yact, m2W_h, m2_b, out);
    kvmn_rowstat<<<BB, 256, 0, stream>>>(out, corr);
    kvmn_logsoftmax<<<dim3((VV / 4 + 255) / 256, BB), 256, 0, stream>>>(out, corr);
}
